// GNNNet_77850577207604
// MI455X (gfx1250) — compile-verified, matrix operand order checked by reading, not run on hardware
//
#include <hip/hip_runtime.h>

// ---------------- problem constants (match reference) ----------------
static const int kB  = 128;
static const int kNM = 100000, kEM = 400000;
static const int kNP = 200000, kEP = 3200000;
static const int kDM = 78, kDP = 54;

typedef __attribute__((ext_vector_type(16))) _Float16 v16h;
typedef __attribute__((ext_vector_type(8)))  _Float16 v8h;
typedef __attribute__((ext_vector_type(8)))  float    v8f;

// ---------------- elementwise helpers ----------------
__global__ void zero_kernel(float* __restrict__ p, int n) {
    int i = blockIdx.x * blockDim.x + threadIdx.x;
    if (i < n) p[i] = 0.0f;
}

// emb = alpha * x ; h = x
__global__ void init_prop_kernel(const float* __restrict__ x, float* __restrict__ h,
                                 float* __restrict__ emb, float alpha, int n) {
    int i = blockIdx.x * blockDim.x + threadIdx.x;
    if (i < n) { float v = x[i]; h[i] = v; emb[i] = alpha * v; }
}

// emb += c * h
__global__ void axpy_kernel(float* __restrict__ emb, const float* __restrict__ h,
                            float c, int n) {
    int i = blockIdx.x * blockDim.x + threadIdx.x;
    if (i < n) emb[i] += c * h[i];
}

// ---------------- graph scatter (segment-sum over edges) ----------------
// dst[d[e], :] += src_x[s[e], :]; optional in-degree count. F must be even.
__global__ void scatter_add_kernel(const float* __restrict__ src_x,
                                   const int* __restrict__ src_idx,
                                   const int* __restrict__ dst_idx,
                                   float* __restrict__ dst,
                                   float* __restrict__ deg,
                                   int nedges, int F) {
    int e = blockIdx.x * blockDim.x + threadIdx.x;
    if (e >= nedges) return;
    int s = src_idx[e], d = dst_idx[e];
    const float2* sp = (const float2*)(src_x + (size_t)s * F);
    float*        dp = dst + (size_t)d * F;
    __builtin_prefetch(sp, 0, 0);   // global_prefetch_b8
    if (deg) atomicAdd(&deg[d], 1.0f);
    int F2 = F >> 1;
    for (int f = 0; f < F2; ++f) {
        float2 v = sp[f];
        atomicAdd(&dp[2 * f],     v.x);
        atomicAdd(&dp[2 * f + 1], v.y);
    }
}

// x[r,:] /= max(cnt[r],1)
__global__ void div_cnt_kernel(float* __restrict__ x, const float* __restrict__ cnt,
                               int rows, int F) {
    int i = blockIdx.x * blockDim.x + threadIdx.x;
    if (i >= rows * F) return;
    int r = i / F;
    x[i] /= fmaxf(cnt[r], 1.0f);
}

// ---------------- segment-mean pooling over batch vector ----------------
__global__ void seg_pool_kernel(const float* __restrict__ x, const int* __restrict__ seg,
                                float* __restrict__ sums, float* __restrict__ cnt,
                                int n, int F) {
    int i = blockIdx.x * blockDim.x + threadIdx.x;
    if (i >= n) return;
    int b = seg[i];
    atomicAdd(&cnt[b], 1.0f);
    const float* xp = x + (size_t)i * F;
    float*       sp = sums + (size_t)b * F;
    for (int f = 0; f < F; ++f) atomicAdd(&sp[f], xp[f]);
}

// ---------------- pack f32 [rows,K] -> zero-padded f16 [rowsPad,Kpad] ------
__global__ void pack_f16_kernel(const float* __restrict__ src, int rows, int K,
                                _Float16* __restrict__ dst, int Kpad, int totalPad) {
    int i = blockIdx.x * blockDim.x + threadIdx.x;
    if (i >= totalPad) return;
    int r = i / Kpad, k = i - r * Kpad;
    float v = (r < rows && k < K) ? src[(size_t)r * K + k] : 0.0f;
    dst[i] = (_Float16)v;
}

// concat two f32 blocks directly into packed f16 [rows, Fa+Fb]
__global__ void concat_pack_kernel(const float* __restrict__ a, int Fa,
                                   const float* __restrict__ b, int Fb,
                                   _Float16* __restrict__ out, int rows) {
    int i = blockIdx.x * blockDim.x + threadIdx.x;
    int F = Fa + Fb;
    if (i >= rows * F) return;
    int r = i / F, cx = i - r * F;
    float v = (cx < Fa) ? a[(size_t)r * Fa + cx] : b[(size_t)r * Fb + (cx - Fa)];
    out[i] = (_Float16)v;
}

// ---------------- WMMA GEMM: D = relu?(A1@W1^T [+ A2@W2^T] + bias) --------
// Operands pre-packed f16: A [M, Kpad] (M mult of 16), W [Npad, Kpad], zero-
// padded. Per 32-wide K step, each lane needs halves {khi..khi+7} and
// {khi+16..khi+23} of its row -> two contiguous b128 loads, branch-free.
// One wave computes one 16x16 tile; f32 accumulation; EXEC all-ones at wmma.
__global__ void wmma_gemm_pk_kernel(const _Float16* __restrict__ A1,
                                    const _Float16* __restrict__ W1, int K1,
                                    const _Float16* __restrict__ A2,
                                    const _Float16* __restrict__ W2, int K2,
                                    const float* __restrict__ bias,
                                    float* __restrict__ out,
                                    int M, int N, int relu) {
    const int lane = threadIdx.x;            // 32 lanes = 1 wave (wave32)
    const int nt = blockIdx.x, mt = blockIdx.y;
    const int l15 = lane & 15;
    const int khi = (lane & 16) ? 8 : 0;
    const _Float16* a1 = A1 + (size_t)(mt * 16 + l15) * K1 + khi;
    const _Float16* w1 = W1 + (size_t)(nt * 16 + l15) * K1 + khi;
    v8f c = {};
    for (int k0 = 0; k0 < K1; k0 += 32) {
        v8h alo = *(const v8h*)(a1 + k0);
        v8h ahi = *(const v8h*)(a1 + k0 + 16);
        v8h blo = *(const v8h*)(w1 + k0);
        v8h bhi = *(const v8h*)(w1 + k0 + 16);
        v16h a = __builtin_shufflevector(alo, ahi, 0,1,2,3,4,5,6,7,8,9,10,11,12,13,14,15);
        v16h b = __builtin_shufflevector(blo, bhi, 0,1,2,3,4,5,6,7,8,9,10,11,12,13,14,15);
        c = __builtin_amdgcn_wmma_f32_16x16x32_f16(false, a, false, b, (short)0, c, false, false);
    }
    if (A2) {
        const _Float16* a2 = A2 + (size_t)(mt * 16 + l15) * K2 + khi;
        const _Float16* w2 = W2 + (size_t)(nt * 16 + l15) * K2 + khi;
        for (int k0 = 0; k0 < K2; k0 += 32) {
            v8h alo = *(const v8h*)(a2 + k0);
            v8h ahi = *(const v8h*)(a2 + k0 + 16);
            v8h blo = *(const v8h*)(w2 + k0);
            v8h bhi = *(const v8h*)(w2 + k0 + 16);
            v16h a = __builtin_shufflevector(alo, ahi, 0,1,2,3,4,5,6,7,8,9,10,11,12,13,14,15);
            v16h b = __builtin_shufflevector(blo, bhi, 0,1,2,3,4,5,6,7,8,9,10,11,12,13,14,15);
            c = __builtin_amdgcn_wmma_f32_16x16x32_f16(false, a, false, b, (short)0, c, false, false);
        }
    }
    const int col   = nt * 16 + l15;
    const int rbase = mt * 16 + ((lane & 16) ? 8 : 0);
    if (col < N) {
        float bv = bias ? bias[col] : 0.0f;
#pragma unroll
        for (int r = 0; r < 8; ++r) {
            float v = c[r] + bv;
            if (relu) v = fmaxf(v, 0.0f);
            out[(size_t)(rbase + r) * N + col] = v;   // M always mult of 16
        }
    }
}

// ---------------- final 512->1 layer ----------------
__global__ void final_out_kernel(const float* __restrict__ x, const float* __restrict__ w,
                                 const float* __restrict__ bias, float* __restrict__ out,
                                 int rows, int K) {
    int b = blockIdx.x * blockDim.x + threadIdx.x;
    if (b >= rows) return;
    float acc = bias[0];
    const float* xp = x + (size_t)b * K;
    for (int k = 0; k < K; ++k) acc += xp[k] * w[k];
    out[b] = acc;
}

// ---------------- host orchestration ----------------
static inline int cdiv(int a, int b) { return (a + b - 1) / b; }

extern "C" void kernel_launch(void* const* d_in, const int* in_sizes, int n_in,
                              void* d_out, int out_size, void* d_ws, size_t ws_size,
                              hipStream_t stream) {
    (void)in_sizes; (void)n_in; (void)out_size; (void)ws_size;
    const float* mol_x   = (const float*)d_in[0];
    const int*   mol_ei  = (const int*)  d_in[1];   // [2, EM]: row=ei[0], col=ei[1]
    const int*   mol_bat = (const int*)  d_in[2];
    const float* pro_x   = (const float*)d_in[3];
    const int*   pro_ei  = (const int*)  d_in[4];   // [2, EP]
    const int*   pro_bat = (const int*)  d_in[5];
    const float* mol_W1  = (const float*)d_in[6],  *mol_b1 = (const float*)d_in[7];
    const float* mol_W2  = (const float*)d_in[8],  *mol_b2 = (const float*)d_in[9];
    const float* c1_Wl   = (const float*)d_in[10], *c1_bl  = (const float*)d_in[11];
    const float* c1_Wr   = (const float*)d_in[12];
    const float* c2_Wl   = (const float*)d_in[13], *c2_bl  = (const float*)d_in[14];
    const float* c2_Wr   = (const float*)d_in[15];
    const float* pro_W1  = (const float*)d_in[16], *pro_b1 = (const float*)d_in[17];
    const float* pro_W2  = (const float*)d_in[18], *pro_b2 = (const float*)d_in[19];
    const float* fc1_W   = (const float*)d_in[20], *fc1_b  = (const float*)d_in[21];
    const float* fc2_W   = (const float*)d_in[22], *fc2_b  = (const float*)d_in[23];
    const float* out_W   = (const float*)d_in[24], *out_b  = (const float*)d_in[25];
    float* dout = (float*)d_out;

    float* ws = (float*)d_ws;
    // ---- big region (phase-shared) ----
    float* h0   = ws;                                  // NM*DM   (mol phase)
    float* h1   = ws + (size_t)kNM * kDM;
    float* emb  = ws + (size_t)2 * kNM * kDM;
    float* aggP = ws;                                  // NP*54   (pro phase reuse)
    float* xt1  = ws + (size_t)kNP * kDP;              // NP*54
    float* xt2  = ws + (size_t)2 * kNP * kDP;          // NP*108
    float* degP = ws + (size_t)4 * kNP * kDP;          // NP
    _Float16* Ap1 = (_Float16*)(ws + 43400000);        // NP*64 f16 operand pack
    _Float16* Ap2 = (_Float16*)(ws + 49800000);        // NP*64 f16 operand pack
    // ---- small region ----
    size_t off = 56200000;
    auto falloc = [&](size_t nfl) { float* p = ws + off; off += (nfl + 63) & ~(size_t)63; return p; };
    auto halloc = [&](size_t nh)  { _Float16* p = (_Float16*)(ws + off); off += ((nh + 1) / 2 + 63) & ~(size_t)63; return p; };
    float* cnt    = falloc(kB);
    float* pool   = falloc((size_t)kB * 108);
    float* xh     = falloc((size_t)kB * 256);
    float* x112   = falloc((size_t)kB * 112);
    float* proh   = falloc((size_t)kB * 256);
    float* xt144  = falloc((size_t)kB * 144);
    float* f1     = falloc((size_t)kB * 1024);
    float* f2     = falloc((size_t)kB * 512);
    _Float16* poolPk = halloc((size_t)kB * 128);       // Kpad up to 128
    _Float16* xhPk   = halloc((size_t)kB * 256);
    _Float16* prohPk = halloc((size_t)kB * 256);
    _Float16* xcPk   = halloc((size_t)kB * 256);
    _Float16* f1Pk   = halloc((size_t)kB * 1024);
    _Float16* molW1p = halloc(256 * 96);
    _Float16* molW2p = halloc(112 * 256);
    _Float16* c1Wlp  = halloc(64 * 64);
    _Float16* c1Wrp  = halloc(64 * 64);
    _Float16* c2Wlp  = halloc(112 * 64);
    _Float16* c2Wrp  = halloc(112 * 64);
    _Float16* proW1p = halloc(256 * 128);
    _Float16* proW2p = halloc(144 * 256);
    _Float16* fc1Wp  = halloc(1024 * 256);
    _Float16* fc2Wp  = halloc(512 * 1024);

    const int T = 256;
    auto pack = [&](const float* src, int rows, int K, _Float16* dst, int Kpad, int rowsPad) {
        int tot = rowsPad * Kpad;
        pack_f16_kernel<<<cdiv(tot, T), T, 0, stream>>>(src, rows, K, dst, Kpad, tot);
    };
    // ---- pack all weights (tiny) ----
    pack(mol_W1, 256, 78, molW1p, 96, 256);
    pack(mol_W2, 112, 256, molW2p, 256, 112);
    pack(c1_Wl, 54, 54, c1Wlp, 64, 64);
    pack(c1_Wr, 54, 54, c1Wrp, 64, 64);
    pack(c2_Wl, 108, 54, c2Wlp, 64, 112);
    pack(c2_Wr, 108, 54, c2Wrp, 64, 112);
    pack(pro_W1, 256, 108, proW1p, 128, 256);
    pack(pro_W2, 144, 256, proW2p, 256, 144);
    pack(fc1_W, 1024, 256, fc1Wp, 256, 1024);
    pack(fc2_W, 512, 1024, fc2Wp, 1024, 512);

    // ================= Phase A: molecule branch =================
    const int nMolF = kNM * kDM;
    init_prop_kernel<<<cdiv(nMolF, T), T, 0, stream>>>(mol_x, h0, emb, 0.05f, nMolF);
    float* hc = h0; float* hn = h1;
    const float coef = (1.0f - 0.05f) / 4.0f;
    for (int it = 0; it < 4; ++it) {
        zero_kernel<<<cdiv(nMolF, T), T, 0, stream>>>(hn, nMolF);
        // h = segsum(h[col], row):  src=ei[1], dst=ei[0]
        scatter_add_kernel<<<cdiv(kEM, T), T, 0, stream>>>(hc, mol_ei + kEM, mol_ei,
                                                           hn, nullptr, kEM, kDM);
        axpy_kernel<<<cdiv(nMolF, T), T, 0, stream>>>(emb, hn, coef, nMolF);
        float* t = hc; hc = hn; hn = t;
    }
    zero_kernel<<<cdiv(kB * kDM, T), T, 0, stream>>>(pool, kB * kDM);
    zero_kernel<<<1, kB, 0, stream>>>(cnt, kB);
    seg_pool_kernel<<<cdiv(kNM, T), T, 0, stream>>>(emb, mol_bat, pool, cnt, kNM, kDM);
    div_cnt_kernel<<<cdiv(kB * kDM, T), T, 0, stream>>>(pool, cnt, kB, kDM);
    // mol MLP: 78 -> 256 (relu) -> 112
    pack(pool, kB, 78, poolPk, 96, kB);
    wmma_gemm_pk_kernel<<<dim3(256 / 16, kB / 16), 32, 0, stream>>>(
        poolPk, molW1p, 96, nullptr, nullptr, 0, mol_b1, xh, kB, 256, 1);
    pack(xh, kB, 256, xhPk, 256, kB);
    wmma_gemm_pk_kernel<<<dim3(112 / 16, kB / 16), 32, 0, stream>>>(
        xhPk, molW2p, 256, nullptr, nullptr, 0, mol_b2, x112, kB, 112, 0);

    // ================= Phase B: protein branch =================
    const int mtP = kNP / 16;   // 12500
    // --- SAGE layer 1: xt1 = relu(mean_agg(pro_x)@Wl^T + bl + pro_x@Wr^T) ---
    zero_kernel<<<cdiv(kNP * kDP, T), T, 0, stream>>>(aggP, kNP * kDP);
    zero_kernel<<<cdiv(kNP, T), T, 0, stream>>>(degP, kNP);
    scatter_add_kernel<<<cdiv(kEP, T), T, 0, stream>>>(pro_x, pro_ei, pro_ei + kEP,
                                                       aggP, degP, kEP, kDP);
    div_cnt_kernel<<<cdiv(kNP * kDP, T), T, 0, stream>>>(aggP, degP, kNP, kDP);
    pack(aggP, kNP, kDP, Ap1, 64, kNP);
    pack(pro_x, kNP, kDP, Ap2, 64, kNP);
    wmma_gemm_pk_kernel<<<dim3(4, mtP), 32, 0, stream>>>(
        Ap1, c1Wlp, 64, Ap2, c1Wrp, 64, c1_bl, xt1, kNP, kDP, 1);
    // --- SAGE layer 2: xt2 = relu(mean_agg(xt1)@Wl^T + bl + xt1@Wr^T), N=108 ---
    zero_kernel<<<cdiv(kNP * kDP, T), T, 0, stream>>>(aggP, kNP * kDP);
    zero_kernel<<<cdiv(kNP, T), T, 0, stream>>>(degP, kNP);
    scatter_add_kernel<<<cdiv(kEP, T), T, 0, stream>>>(xt1, pro_ei, pro_ei + kEP,
                                                       aggP, degP, kEP, kDP);
    div_cnt_kernel<<<cdiv(kNP * kDP, T), T, 0, stream>>>(aggP, degP, kNP, kDP);
    pack(aggP, kNP, kDP, Ap1, 64, kNP);
    pack(xt1,  kNP, kDP, Ap2, 64, kNP);
    wmma_gemm_pk_kernel<<<dim3(cdiv(108, 16), mtP), 32, 0, stream>>>(
        Ap1, c2Wlp, 64, Ap2, c2Wrp, 64, c2_bl, xt2, kNP, 108, 1);
    // --- pool + pro MLP: 108 -> 256 (relu) -> 144 ---
    zero_kernel<<<cdiv(kB * 108, T), T, 0, stream>>>(pool, kB * 108);
    zero_kernel<<<1, kB, 0, stream>>>(cnt, kB);
    seg_pool_kernel<<<cdiv(kNP, T), T, 0, stream>>>(xt2, pro_bat, pool, cnt, kNP, 108);
    div_cnt_kernel<<<cdiv(kB * 108, T), T, 0, stream>>>(pool, cnt, kB, 108);
    pack(pool, kB, 108, poolPk, 128, kB);
    wmma_gemm_pk_kernel<<<dim3(256 / 16, kB / 16), 32, 0, stream>>>(
        poolPk, proW1p, 128, nullptr, nullptr, 0, pro_b1, proh, kB, 256, 1);
    pack(proh, kB, 256, prohPk, 256, kB);
    wmma_gemm_pk_kernel<<<dim3(144 / 16, kB / 16), 32, 0, stream>>>(
        prohPk, proW2p, 256, nullptr, nullptr, 0, pro_b2, xt144, kB, 144, 0);

    // ================= Phase C: head =================
    concat_pack_kernel<<<cdiv(kB * 256, T), T, 0, stream>>>(x112, 112, xt144, 144, xcPk, kB);
    wmma_gemm_pk_kernel<<<dim3(1024 / 16, kB / 16), 32, 0, stream>>>(
        xcPk, fc1Wp, 256, nullptr, nullptr, 0, fc1_b, f1, kB, 1024, 1);
    pack(f1, kB, 1024, f1Pk, 1024, kB);
    wmma_gemm_pk_kernel<<<dim3(512 / 16, kB / 16), 32, 0, stream>>>(
        f1Pk, fc2Wp, 1024, nullptr, nullptr, 0, fc2_b, f2, kB, 512, 1);
    final_out_kernel<<<1, kB, 0, stream>>>(f2, out_W, out_b, dout, kB, 512);
}
